// MlpMoeBlock_13048110645659
// MI455X (gfx1250) — matrix operand order repaired and checked by reading, not compiled, not run on hardware
//
#include <hip/hip_runtime.h>
#include <hip/hip_bf16.h>
#include <math.h>

// ---------------------------------------------------------------------------
// MoE MLP block for MI455X (gfx1250, wave32, WMMA + TDM).
//   tokens N = 16384, H = 1024, M = 4096, E = 8, top-2 routing.
// HBM-bound on the one-time f32 weight read (268 MB) -> convert f32->bf16
// while staging into LDS, compute with v_wmma_f32_16x16x32_bf16 (f32 acc).
// A tiles (bf16) are DMA'd into LDS by the Tensor Data Mover
// (tensor_load_to_lds + s_wait_tensorcnt), with TDM row padding for
// bank-conflict-free ds_load_b128 fragment reads.
// hmid staged per (expert, 4096-token slab) in bf16 (33.5 MB, L2-resident).
// ---------------------------------------------------------------------------

typedef __attribute__((ext_vector_type(8)))  __bf16 v8bf;
typedef __attribute__((ext_vector_type(16))) __bf16 v16bf;
typedef __attribute__((ext_vector_type(8)))  float  v8f;
typedef __attribute__((ext_vector_type(4))) unsigned int u32x4;
typedef __attribute__((ext_vector_type(8))) int          i32x8;
typedef __attribute__((ext_vector_type(4))) int          i32x4;

#define NSEQ   8
#define LSEQ   2048
#define HDIM   1024
#define NTOK   (NSEQ * LSEQ)       // 16384
#define MDIM   4096
#define NEXP   8
#define SLAB   4096                // tokens per slab
#define NSLAB  (NTOK / SLAB)       // 4
#define EPSF   1e-9f

// LDS A-tile row stride in bf16 elems: 32 data + 8 pad (TDM pad) = 40 (80 B)
#define SA_STRIDE 40

__device__ __forceinline__ __bf16 f2bf(float f) {
  unsigned u = __builtin_bit_cast(unsigned, f);
  u += 0x7fffu + ((u >> 16) & 1u);            // round-to-nearest-even
  unsigned short h = (unsigned short)(u >> 16);
  return __builtin_bit_cast(__bf16, h);
}

__device__ __forceinline__ float gelu_exact(float v) {
  return 0.5f * v * (1.0f + erff(v * 0.70710678118654752f));
}

// ---------------------------------------------------------------------------
// TDM: DMA one 128(row) x 32(k) bf16 tile from global into LDS.
// D# per CDNA5 ISA ch.8: group0 = {count/flags, lds_addr, global_addr, type},
// group1 = {mask/data_size/pad, tensor dims, tile dims, strides}.
// pad_interval = 16 DWORDs (code 3), pad_amount = 4 DWORDs (code 3):
// every 64 B row gets +16 B -> LDS row stride 80 B (SA_STRIDE bf16).
// This toolchain exposes the 6-arg builtin (g0, g1, g2, g3, extra, cpol).
// ---------------------------------------------------------------------------
__device__ __forceinline__ void tdm_load_a_tile(const __bf16* gsrc,
                                                unsigned lds_addr,
                                                int lda_elems) {
  unsigned long long ga = (unsigned long long)(uintptr_t)gsrc;
  u32x4 g0 = {
      1u,                                               // count = 1 descriptor
      lds_addr,                                         // LDS byte address
      (unsigned)(ga & 0xffffffffull),                   // global_addr[31:0]
      (unsigned)((ga >> 32) & 0x1ffffffull) | (2u << 30) // [56:32] | type=2
  };
  const unsigned dim0 = 0x40000000u;        // huge tensor dims: never OOB
  const unsigned dim1 = 0x40000000u;
  const unsigned long long s0 = (unsigned long long)lda_elems;
  unsigned w0 = (1u << 16)        // data_size = 2 bytes (code 1)
              | (1u << 20)        // pad_enable
              | (3u << 22)        // pad_interval: 16 DWORDs per row
              | (3u << 25);       // pad_amount: 4 DWORDs (16 B)
  unsigned w1 = (dim0 << 16);                         // tensor_dim0[15:0]
  unsigned w2 = (dim0 >> 16) | (dim1 << 16);          // dim0 hi | dim1 lo
  unsigned w3 = (dim1 >> 16) | (32u << 16);           // dim1 hi | tile_dim0=32
  unsigned w4 = 128u;                                 // tile_dim1=128, dim2=0
  unsigned w5 = (unsigned)(s0 & 0xffffffffull);       // tensor_dim0_stride lo
  unsigned w6 = (unsigned)((s0 >> 32) & 0xffffull);   // stride0 hi | stride1=0
  unsigned w7 = 0u;
  i32x8 g1 = { (int)w0, (int)w1, (int)w2, (int)w3,
               (int)w4, (int)w5, (int)w6, (int)w7 };
  i32x4 gz4 = { 0, 0, 0, 0 };
  i32x8 gz8 = { 0, 0, 0, 0, 0, 0, 0, 0 };
  __builtin_amdgcn_tensor_load_to_lds(g0, g1, gz4, gz4, gz8, 0);
}

// ---------------------------------------------------------------------------
// Zero the f32 output accumulator (float4 stores).
// ---------------------------------------------------------------------------
__global__ __launch_bounds__(256) void zero_kernel(float* __restrict__ p) {
  size_t i = ((size_t)blockIdx.x * 256 + threadIdx.x) * 4;
  float4 z; z.x = 0.f; z.y = 0.f; z.z = 0.f; z.w = 0.f;
  *(float4*)(p + i) = z;
}

// ---------------------------------------------------------------------------
// Cast x f32 -> bf16 once (A-operand for GEMM1).
// ---------------------------------------------------------------------------
__global__ __launch_bounds__(256) void cast_x_kernel(const float* __restrict__ x,
                                                     __bf16* __restrict__ xb) {
  size_t i = ((size_t)blockIdx.x * 256 + threadIdx.x) * 4;
  float4 v = *(const float4*)(x + i);
  unsigned short h0 = __builtin_bit_cast(unsigned short, f2bf(v.x));
  unsigned short h1 = __builtin_bit_cast(unsigned short, f2bf(v.y));
  unsigned short h2 = __builtin_bit_cast(unsigned short, f2bf(v.z));
  unsigned short h3 = __builtin_bit_cast(unsigned short, f2bf(v.w));
  uint2 pk;
  pk.x = (unsigned)h0 | ((unsigned)h1 << 16);
  pk.y = (unsigned)h2 | ((unsigned)h3 << 16);
  *(uint2*)(xb + i) = pk;
}

// ---------------------------------------------------------------------------
// Router: logits -> softmax -> top-2 -> normalized combine weights per expert.
// wcomb[tok][e] = (w_e / (w_top1 + w_top2 + eps)), 0 for unselected experts.
// One wave per token; lane-strided dot products + shuffle reduction.
// ---------------------------------------------------------------------------
__global__ __launch_bounds__(128) void router_kernel(const float* __restrict__ x,
                                                     const float* __restrict__ Wr,
                                                     const float* __restrict__ br,
                                                     float* __restrict__ wcomb) {
  const int wid  = threadIdx.x >> 5;
  const int lane = threadIdx.x & 31;
  const int tok  = blockIdx.x * 4 + wid;
  const float* xr = x + (size_t)tok * HDIM;

  float acc[NEXP];
#pragma unroll
  for (int e = 0; e < NEXP; ++e) acc[e] = 0.f;

  for (int h = lane; h < HDIM; h += 32) {
    float xv = xr[h];
    const float* wr = Wr + h * NEXP;
#pragma unroll
    for (int e = 0; e < NEXP; ++e) acc[e] += xv * wr[e];
  }
#pragma unroll
  for (int e = 0; e < NEXP; ++e) {
#pragma unroll
    for (int off = 16; off >= 1; off >>= 1)
      acc[e] += __shfl_xor(acc[e], off, 32);
  }
  if (lane == 0) {
    float lg[NEXP], p[NEXP];
    float mx = -1e30f;
#pragma unroll
    for (int e = 0; e < NEXP; ++e) { lg[e] = acc[e] + br[e]; mx = fmaxf(mx, lg[e]); }
    float s = 0.f;
#pragma unroll
    for (int e = 0; e < NEXP; ++e) { p[e] = __expf(lg[e] - mx); s += p[e]; }
    float inv = 1.f / s;
#pragma unroll
    for (int e = 0; e < NEXP; ++e) p[e] *= inv;
    int i1 = 0; float v1 = p[0];
#pragma unroll
    for (int e = 1; e < NEXP; ++e) if (p[e] > v1) { v1 = p[e]; i1 = e; }
    int i2 = (i1 == 0) ? 1 : 0; float v2 = p[i2];
#pragma unroll
    for (int e = 0; e < NEXP; ++e) if (e != i1 && p[e] > v2) { v2 = p[e]; i2 = e; }
    float denom = v1 + v2 + EPSF;          // sum(vals) + eps
    float w1 = v1 / denom, w2 = v2 / denom;
    float scale = 1.f / (w1 + w2 + EPSF);  // 1/(weight_acc + eps)
    float* wc = wcomb + (size_t)tok * NEXP;
#pragma unroll
    for (int e = 0; e < NEXP; ++e) wc[e] = 0.f;
    wc[i1] = w1 * scale;
    wc[i2] = w2 * scale;
  }
}

// ---------------------------------------------------------------------------
// Tiled WMMA GEMM: C(128x128 block) = A(bf16, MxK) x B(f32 -> bf16, KxN).
// A tiles arrive via TDM (tensor_load_to_lds, double-buffered, wave 0 issues,
// s_wait_tensorcnt drains). B tiles are convert-staged by all threads.
// PHASE1: epilogue = bias + exact GELU, store bf16 to Hout.
// PHASE2: epilogue = bias, then Yout[tok][col] += wcomb[tok][e] * y  (f32).
// 8 waves, 2x4 wave grid, 4x2 16x16 WMMA tiles per wave, K-step 32.
// ---------------------------------------------------------------------------
template <bool PHASE1>
__global__ __launch_bounds__(256)
void moe_gemm(const __bf16* __restrict__ A, int lda,
              const float*  __restrict__ B, int ldb,
              const float*  __restrict__ bias,
              int K, int N,
              __bf16* __restrict__ Hout,
              float*  __restrict__ Yout,
              const float* __restrict__ wcomb,
              int expert, int tok0) {
  __shared__ __bf16 sA[2][128 * SA_STRIDE];  // TDM-filled, stride 40 (80 B)
  __shared__ __bf16 sB[2][128][48];          // [n][k] transposed tile

  const int tid   = threadIdx.x;
  const int m0    = blockIdx.x * 128;
  const int n0    = blockIdx.y * 128;
  const int wid   = tid >> 5;
  const int lane  = tid & 31;
  const int lhalf = lane >> 4;        // 0 or 1
  const int lmod  = lane & 15;
  const int wm    = (wid & 1) * 64;   // wave row offset
  const int wn    = (wid >> 1) * 32;  // wave col offset

  // Wave-relative LDS byte addresses of the two A buffers (generic-pointer
  // low 32 bits are the LDS offset per the CDNA5 flat-aperture mapping).
  const unsigned ldsA0 = (unsigned)(uintptr_t)&sA[0][0];
  const unsigned ldsA1 = (unsigned)(uintptr_t)&sA[1][0];

  v8f acc[4][2];
#pragma unroll
  for (int mi = 0; mi < 4; ++mi)
#pragma unroll
    for (int ni = 0; ni < 2; ++ni)
      acc[mi][ni] = (v8f){0.f, 0.f, 0.f, 0.f, 0.f, 0.f, 0.f, 0.f};

  // B tile: 32 k x 128 n f32, convert -> bf16, store transposed [n][k].
  auto stageB = [&](int buf, int k0) {
#pragma unroll
    for (int it = 0; it < 4; ++it) {
      int c  = tid + it * 256;       // 0..1023
      int k  = c >> 5;               // 0..31
      int n4 = (c & 31) * 4;
      float4 v = *(const float4*)(B + (size_t)(k0 + k) * ldb + n0 + n4);
      sB[buf][n4 + 0][k] = f2bf(v.x);
      sB[buf][n4 + 1][k] = f2bf(v.y);
      sB[buf][n4 + 2][k] = f2bf(v.z);
      sB[buf][n4 + 3][k] = f2bf(v.w);
    }
  };

  // Prologue: DMA A tile 0, convert-stage B tile 0.
  if (tid < 32)
    tdm_load_a_tile(A + (size_t)m0 * lda, ldsA0, lda);
  stageB(0, 0);
  if (tid < 32)
    __builtin_amdgcn_s_wait_tensorcnt(0);
  __syncthreads();

  for (int k0 = 0; k0 < K; k0 += 32) {
    const int buf = (k0 >> 5) & 1;
    if (k0 + 32 < K) {
      if (k0 + 64 < K)   // hint next-next B tile toward L2 (global_prefetch_b8)
        __builtin_prefetch(B + (size_t)(k0 + 64) * ldb + n0 + (lane << 2), 0, 1);
      if (tid < 32)      // TDM next A tile into the other buffer
        tdm_load_a_tile(A + (size_t)m0 * lda + (k0 + 32),
                        buf ? ldsA0 : ldsA1, lda);
      stageB(buf ^ 1, k0 + 32);
    }

    const __bf16* sAb = &sA[buf][0];
    // A fragment (16x32 bf16): lane holds row = lmod, K = 16h + 0..15
    v16bf afrag[4];
#pragma unroll
    for (int mi = 0; mi < 4; ++mi) {
      int row = wm + mi * 16 + lmod;
      int b   = lhalf * 16;
      v8bf lo = *(const v8bf*)(sAb + row * SA_STRIDE + b);
      v8bf hi = *(const v8bf*)(sAb + row * SA_STRIDE + b + 8);
      afrag[mi] = __builtin_shufflevector(lo, hi, 0, 1, 2, 3, 4, 5, 6, 7,
                                          8, 9, 10, 11, 12, 13, 14, 15);
    }
    // B fragment (32x16 bf16): lane holds col = lmod, K = 16h + 0..15
    v16bf bfrag[2];
#pragma unroll
    for (int ni = 0; ni < 2; ++ni) {
      int col  = wn + ni * 16 + lmod;
      int koff = lhalf * 16;
      v8bf lo = *(const v8bf*)&sB[buf][col][koff];
      v8bf hi = *(const v8bf*)&sB[buf][col][koff + 8];
      bfrag[ni] = __builtin_shufflevector(lo, hi, 0, 1, 2, 3, 4, 5, 6, 7,
                                          8, 9, 10, 11, 12, 13, 14, 15);
    }
#pragma unroll
    for (int mi = 0; mi < 4; ++mi)
#pragma unroll
      for (int ni = 0; ni < 2; ++ni)
        acc[mi][ni] = __builtin_amdgcn_wmma_f32_16x16x32_bf16(
            false, afrag[mi], false, bfrag[ni], (short)0, acc[mi][ni],
            false, false);

    if (tid < 32)        // drain the in-flight TDM before the barrier
      __builtin_amdgcn_s_wait_tensorcnt(0);
    __syncthreads();
  }

  // Epilogue. D layout: lane lmod = N col, VGPR i -> M = 8*lhalf + i.
  if (PHASE1) {
#pragma unroll
    for (int mi = 0; mi < 4; ++mi)
#pragma unroll
      for (int ni = 0; ni < 2; ++ni) {
        int col  = n0 + wn + ni * 16 + lmod;
        float bv = bias[col];
#pragma unroll
        for (int i = 0; i < 8; ++i) {
          int row = m0 + wm + mi * 16 + lhalf * 8 + i;
          float v = acc[mi][ni][i] + bv;
          Hout[(size_t)row * N + col] = f2bf(gelu_exact(v));
        }
      }
  } else {
#pragma unroll
    for (int mi = 0; mi < 4; ++mi)
#pragma unroll
      for (int ni = 0; ni < 2; ++ni) {
        int col  = n0 + wn + ni * 16 + lmod;
        float bv = bias[col];
#pragma unroll
        for (int i = 0; i < 8; ++i) {
          int row  = m0 + wm + mi * 16 + lhalf * 8 + i;
          int tok  = tok0 + row;
          float wc = wcomb[(size_t)tok * NEXP + expert];
          float y  = acc[mi][ni][i] + bv;
          float* o = Yout + (size_t)tok * HDIM + col;
          *o += wc * y;   // exclusive owner; kernels serialize on stream
        }
      }
  }
}

// ---------------------------------------------------------------------------
// Launch: zero-out, cast x, router, then 8 experts x 4 slabs x (GEMM1, GEMM2).
// Workspace: Xbf16 (32 MB) | Hs slab bf16 (32 MB) | wcomb (512 KB) ~= 67.6 MB.
// ---------------------------------------------------------------------------
extern "C" void kernel_launch(void* const* d_in, const int* in_sizes, int n_in,
                              void* d_out, int out_size, void* d_ws, size_t ws_size,
                              hipStream_t stream) {
  const float* x  = (const float*)d_in[0];
  const float* Wr = (const float*)d_in[1];
  const float* br = (const float*)d_in[2];
  const float* W1 = (const float*)d_in[3];
  const float* b1 = (const float*)d_in[4];
  const float* W2 = (const float*)d_in[5];
  const float* b2 = (const float*)d_in[6];
  float* out = (float*)d_out;

  char* ws = (char*)d_ws;
  __bf16* Xb    = (__bf16*)ws;
  __bf16* Hs    = (__bf16*)(ws + (size_t)NTOK * HDIM * 2);
  float*  wcomb = (float*)(ws + (size_t)NTOK * HDIM * 2 + (size_t)SLAB * MDIM * 2);

  const int nelem = NTOK * HDIM;                       // 16,777,216
  zero_kernel<<<nelem / 1024, 256, 0, stream>>>(out);
  cast_x_kernel<<<nelem / 1024, 256, 0, stream>>>(x, Xb);
  router_kernel<<<NTOK / 4, 128, 0, stream>>>(x, Wr, br, wcomb);

  dim3 blk(256);
  for (int e = 0; e < NEXP; ++e) {
    for (int s = 0; s < NSLAB; ++s) {
      const int tok0 = s * SLAB;
      dim3 g1(SLAB / 128, MDIM / 128);                 // 32 x 32
      moe_gemm<true><<<g1, blk, 0, stream>>>(
          Xb + (size_t)tok0 * HDIM, HDIM,
          W1 + (size_t)e * HDIM * MDIM, MDIM,
          b1 + (size_t)e * MDIM,
          HDIM, MDIM, Hs, nullptr, nullptr, e, tok0);
      dim3 g2(SLAB / 128, HDIM / 128);                 // 32 x 8
      moe_gemm<false><<<g2, blk, 0, stream>>>(
          Hs, MDIM,
          W2 + (size_t)e * MDIM * HDIM, HDIM,
          b2 + (size_t)e * HDIM,
          MDIM, HDIM, nullptr, out, wcomb, e, tok0);
    }
  }
}
